// Diff_SSM_Block_68822555951832
// MI455X (gfx1250) — compile-verified, matrix operand order checked by reading, not run on hardware
//
#include <hip/hip_runtime.h>
#include <hip/hip_bf16.h>
#include <cstdint>

// ---- problem dims (fixed by reference) ----
#define H_   1024
#define B_   8
#define L_   4096
#define DI_  2048
#define DS_  16
#define DC_  4
#define DR_  64
#define HR_  256
#define NTOK (B_ * L_)      // 32768 tokens
#define NC_  16             // scan chunks
#define CL_  (L_ / NC_)     // 256 steps per chunk

typedef __attribute__((ext_vector_type(16))) _Float16 v16h;
typedef __attribute__((ext_vector_type(8)))  float    v8f;

union FragU { v16h v; uint4 q[2]; };

// ---- WMMA GEMM tiling ----
#define BM 128
#define BN 128
#define BK 32
#define LDSP 40   // padded LDS row stride in halves (80B -> conflict-free b128 reads)

__device__ __forceinline__ float siluf(float v) { return v / (1.f + __expf(-v)); }

// Load one 16x32 f16 fragment from an LDS tile row-base pointer (already offset
// by 8*(lane>>4) halves).  e0..7 <- K = 8*hi + 0..7 ; e8..15 <- K = 16 + 8*hi + 0..7
__device__ __forceinline__ v16h ld_frag(const _Float16* p) {
  FragU f;
  f.q[0] = *(const uint4*)(p);
  f.q[1] = *(const uint4*)(p + 16);
  return f.v;
}

enum {
  EPI_F16 = 0,        // store f16
  EPI_BIAS_F16,       // +bias, store f16
  EPI_BIAS_SILU_F16,  // +bias, silu, store f16
  EPI_F32,            // store f32
  EPI_BIAS_SP_F16,    // +bias, softplus, store f16
  EPI_FINAL           // +bias, out = resid + alpha[b]*v, store f32
};

// C[M,N] = A[M,K] * Bw[N,K]^T   (A row-major f16, Bw row-major f16 = weight)
// Block tile 128x128, 8 waves (2x4), each wave 64x32 = 4x2 WMMA tiles.
// K-loop is software-pipelined with register double buffering: the next tile's
// global_load_b128s are issued right after the barrier and only consumed (and
// waited on) when stored to LDS in the next iteration, so their latency is
// hidden under the 8 WMMAs + fragment ds_loads of the current tile.
// GUARD=true only for the single N=96 (xproj) launch; all other N are
// multiples of BN so the guard (and its register pressure) compiles out.
template<int EPI, bool GUARD>
__global__ __launch_bounds__(256) void gemm_f16(
    const _Float16* __restrict__ A, int lda,
    const _Float16* __restrict__ Bw, int ldb,
    int N, int K, int ldc,
    float* __restrict__ outF, _Float16* __restrict__ outH,
    const float* __restrict__ bias,
    const float* __restrict__ resid, const float* __restrict__ modv)
{
  __shared__ _Float16 As[BM * LDSP];
  __shared__ _Float16 Bs[BN * LDSP];
  const int tid  = threadIdx.x;
  const int lane = tid & 31;
  const int wid  = tid >> 5;
  const int wm   = wid >> 2;      // 0..1 -> 64 rows per wave
  const int wn   = wid & 3;       // 0..3 -> 32 cols per wave
  const int bm   = blockIdx.x * BM;
  const int bn   = blockIdx.y * BN;

  // staging map: each thread owns rows lr and lr+64, K-chunk lc (8 halves)
  const int lr = tid >> 2;            // 0..63
  const int lc = (tid & 3) * 8;       // 0,8,16,24
  const _Float16* aP0 = A + (size_t)(bm + lr) * lda + lc;
  const _Float16* aP1 = aP0 + (size_t)64 * lda;
  const int n0 = bn + lr, n1 = n0 + 64;
  const bool bv0 = !GUARD || (n0 < N);
  const bool bv1 = !GUARD || (n1 < N);
  const _Float16* bP0 = Bw + (size_t)n0 * ldb + lc;
  const _Float16* bP1 = Bw + (size_t)n1 * ldb + lc;
  _Float16* sA0 = &As[lr * LDSP + lc];
  _Float16* sA1 = &As[(lr + 64) * LDSP + lc];
  _Float16* sB0 = &Bs[lr * LDSP + lc];
  _Float16* sB1 = &Bs[(lr + 64) * LDSP + lc];

  v8f acc[4][2];
  const v8f vz = {0.f,0.f,0.f,0.f,0.f,0.f,0.f,0.f};
  for (int i = 0; i < 4; ++i)
    for (int j = 0; j < 2; ++j) acc[i][j] = vz;

  const uint4 uz = {0u, 0u, 0u, 0u};
  uint4 ra0, ra1, rb0, rb1;
  // prologue: load K-tile 0 into registers
  ra0 = *(const uint4*)(aP0);
  ra1 = *(const uint4*)(aP1);
  if (GUARD) {
    rb0 = bv0 ? *(const uint4*)(bP0) : uz;
    rb1 = bv1 ? *(const uint4*)(bP1) : uz;
  } else {
    rb0 = *(const uint4*)(bP0);
    rb1 = *(const uint4*)(bP1);
  }

  const int fr = lane & 15;
  const int fh = (lane >> 4) * 8;

  for (int k0 = 0; k0 < K; k0 += BK) {
    // commit staged registers to LDS (implicit wait on the in-flight loads)
    *(uint4*)sA0 = ra0;
    *(uint4*)sA1 = ra1;
    *(uint4*)sB0 = rb0;
    *(uint4*)sB1 = rb1;
    __syncthreads();
    // kick off next tile's loads; latency hidden by the WMMA phase below
    if (k0 + BK < K) {
      ra0 = *(const uint4*)(aP0 + k0 + BK);
      ra1 = *(const uint4*)(aP1 + k0 + BK);
      if (GUARD) {
        rb0 = bv0 ? *(const uint4*)(bP0 + k0 + BK) : uz;
        rb1 = bv1 ? *(const uint4*)(bP1 + k0 + BK) : uz;
      } else {
        rb0 = *(const uint4*)(bP0 + k0 + BK);
        rb1 = *(const uint4*)(bP1 + k0 + BK);
      }
    }

    v16h bf[2];
#pragma unroll
    for (int j = 0; j < 2; ++j)
      bf[j] = ld_frag(&Bs[(wn * 32 + j * 16 + fr) * LDSP + fh]);
#pragma unroll
    for (int i = 0; i < 4; ++i) {
      v16h af = ld_frag(&As[(wm * 64 + i * 16 + fr) * LDSP + fh]);
#pragma unroll
      for (int j = 0; j < 2; ++j)
        acc[i][j] = __builtin_amdgcn_wmma_f32_16x16x32_f16(
            false, af, false, bf[j], (short)0, acc[i][j], false, false);
    }
    __syncthreads();
  }

  const int cl = lane & 15, hi = lane >> 4;
#pragma unroll
  for (int i = 0; i < 4; ++i)
#pragma unroll
    for (int j = 0; j < 2; ++j)
#pragma unroll
      for (int r = 0; r < 8; ++r) {
        int row = bm + wm * 64 + i * 16 + r + 8 * hi;
        int col = bn + wn * 32 + j * 16 + cl;
        if (GUARD && col >= N) continue;
        float v = acc[i][j][r];
        if (EPI == EPI_BIAS_F16 || EPI == EPI_BIAS_SILU_F16 ||
            EPI == EPI_BIAS_SP_F16 || EPI == EPI_FINAL)
          v += bias[col];
        if (EPI == EPI_BIAS_SILU_F16) v = siluf(v);
        if (EPI == EPI_BIAS_SP_F16) v = (v > 20.f) ? v : log1pf(__expf(v));
        size_t o = (size_t)row * ldc + col;
        if (EPI == EPI_F32) {
          outF[o] = v;
        } else if (EPI == EPI_FINAL) {
          int b = row >> 12;  // row / L_
          outF[o] = resid[o] + modv[b * 3 * H_ + 2 * H_ + col] * v;
        } else {
          outH[o] = (_Float16)v;
        }
      }
}

// ---- small / elementwise kernels ----

__global__ void k_cvt16(const float* __restrict__ in, _Float16* __restrict__ out, int n) {
  int i = blockIdx.x * 256 + threadIdx.x;
  if (i < n) out[i] = (_Float16)in[i];
}

__global__ void k_negexp(const float* __restrict__ a, float* __restrict__ o, int n) {
  int i = blockIdx.x * 256 + threadIdx.x;
  if (i < n) o[i] = -__expf(a[i]);
}

// mod = silu(c) @ adaln_w^T + adaln_b      [B, 3H]
__global__ __launch_bounds__(256) void k_mod(const float* __restrict__ c,
                                             const float* __restrict__ w,
                                             const float* __restrict__ bias,
                                             float* __restrict__ mod) {
  __shared__ float cs[H_];
  int b = blockIdx.y;
  int o = blockIdx.x * 256 + threadIdx.x;
  for (int i = threadIdx.x; i < H_; i += 256) {
    float v = c[b * H_ + i];
    cs[i] = siluf(v);
  }
  __syncthreads();
  const float* wr = w + (size_t)o * H_;
  float acc = 0.f;
  for (int i = 0; i < H_; ++i) acc += cs[i] * wr[i];
  mod[b * 3 * H_ + o] = acc + bias[o];
}

// LayerNorm (eps 1e-6, no affine) + adaLN modulate -> x1 (f16)
__global__ __launch_bounds__(256) void k_ln_mod(const float* __restrict__ x,
                                                const float* __restrict__ mod,
                                                _Float16* __restrict__ x1h) {
  __shared__ float r1[256], r2[256];
  int row = blockIdx.x;
  int b = row >> 12;  // row / L_
  const float* xr = x + (size_t)row * H_;
  float vals[4], s = 0.f, ss = 0.f;
#pragma unroll
  for (int i = 0; i < 4; ++i) {
    float v = xr[threadIdx.x + 256 * i];
    vals[i] = v; s += v; ss += v * v;
  }
  r1[threadIdx.x] = s; r2[threadIdx.x] = ss;
  __syncthreads();
  for (int st = 128; st > 0; st >>= 1) {
    if (threadIdx.x < st) { r1[threadIdx.x] += r1[threadIdx.x + st];
                            r2[threadIdx.x] += r2[threadIdx.x + st]; }
    __syncthreads();
  }
  float mu   = r1[0] * (1.f / H_);
  float var  = r2[0] * (1.f / H_) - mu * mu;
  float rstd = rsqrtf(var + 1e-6f);
  const float* mb = mod + b * 3 * H_;
#pragma unroll
  for (int i = 0; i < 4; ++i) {
    int h = threadIdx.x + 256 * i;
    float y = (vals[i] - mu) * rstd * (1.f + mb[H_ + h]) + mb[h];
    x1h[(size_t)row * H_ + h] = (_Float16)y;
  }
}

// causal depthwise conv1d + SiLU; input = xz[:, :DI] (row stride 2*DI)
__global__ void k_conv(const _Float16* __restrict__ xz,
                       const float* __restrict__ cw,
                       const float* __restrict__ cb,
                       _Float16* __restrict__ xc) {
  int d = blockIdx.x * 256 + threadIdx.x;
  int t = blockIdx.y, b = blockIdx.z;
  float acc = cb[d];
#pragma unroll
  for (int k = 0; k < DC_; ++k) {
    int tt = t - (DC_ - 1) + k;
    if (tt >= 0)
      acc += cw[d * DC_ + k] * (float)xz[(size_t)(b * L_ + tt) * (2 * DI_) + d];
  }
  xc[(size_t)(b * L_ + t) * DI_ + d] = (_Float16)siluf(acc);
}

// extract dt (first DR cols of dbl) to f16
__global__ void k_dth(const float* __restrict__ dbl, _Float16* __restrict__ dth, int n) {
  int i = blockIdx.x * 256 + threadIdx.x;
  if (i >= n) return;
  int row = i >> 6, c = i & 63;
  dth[i] = (_Float16)dbl[(size_t)row * 96 + c];
}

// ---- chunked selective scan ----
// pass A: per-chunk final state with h0=0, plus Sdt = sum(dt)
__global__ __launch_bounds__(256) void k_scan_partial(
    const _Float16* __restrict__ dt16, const _Float16* __restrict__ xc,
    const float* __restrict__ dbl, const float* __restrict__ Aneg,
    float* __restrict__ hfin, float* __restrict__ Sdt)
{
  int d = blockIdx.x * 256 + threadIdx.x;
  int c = blockIdx.y, b = blockIdx.z;
  float A2[DS_], h[DS_];
#pragma unroll
  for (int s = 0; s < DS_; ++s) {
    A2[s] = Aneg[d * DS_ + s] * 1.44269504f;  // A * log2(e)
    h[s] = 0.f;
  }
  float S = 0.f;
  int t0 = c * CL_;
  for (int t = t0; t < t0 + CL_; ++t) {
    size_t rowN = (size_t)(b * L_ + t);
    float dt = (float)dt16[rowN * DI_ + d];
    float u  = dt * (float)xc[rowN * DI_ + d];
    S += dt;
    const float* Bv = dbl + rowN * 96 + DR_;
#pragma unroll
    for (int s = 0; s < DS_; ++s)
      h[s] = h[s] * exp2f(dt * A2[s]) + u * Bv[s];
  }
  size_t o = (size_t)(b * NC_ + c) * DI_ + d;
  Sdt[o] = S;
#pragma unroll
  for (int s = 0; s < DS_; ++s) hfin[o * DS_ + s] = h[s];
}

// pass B: sequential combine over chunks; decays compose: exp(A*Sdt)
__global__ __launch_bounds__(256) void k_scan_combine(
    const float* __restrict__ hfin, const float* __restrict__ Sdt,
    const float* __restrict__ Aneg, float* __restrict__ h0buf)
{
  int d = blockIdx.x * 256 + threadIdx.x;
  int b = blockIdx.y;
  float A2[DS_], h[DS_];
#pragma unroll
  for (int s = 0; s < DS_; ++s) {
    A2[s] = Aneg[d * DS_ + s] * 1.44269504f;
    h[s] = 0.f;
  }
  for (int c = 0; c < NC_; ++c) {
    size_t o = (size_t)(b * NC_ + c) * DI_ + d;
#pragma unroll
    for (int s = 0; s < DS_; ++s) h0buf[o * DS_ + s] = h[s];
    float S = Sdt[o];
#pragma unroll
    for (int s = 0; s < DS_; ++s)
      h[s] = hfin[o * DS_ + s] + exp2f(S * A2[s]) * h[s];
  }
}

// pass C: replay each chunk from its true h0, emit y_t = <h_t, C_t>
__global__ __launch_bounds__(256) void k_scan_apply(
    const _Float16* __restrict__ dt16, const _Float16* __restrict__ xc,
    const float* __restrict__ dbl, const float* __restrict__ Aneg,
    const float* __restrict__ h0buf, float* __restrict__ ys)
{
  int d = blockIdx.x * 256 + threadIdx.x;
  int c = blockIdx.y, b = blockIdx.z;
  float A2[DS_], h[DS_];
  size_t co = (size_t)(b * NC_ + c) * DI_ + d;
#pragma unroll
  for (int s = 0; s < DS_; ++s) {
    A2[s] = Aneg[d * DS_ + s] * 1.44269504f;
    h[s] = h0buf[co * DS_ + s];
  }
  int t0 = c * CL_;
  for (int t = t0; t < t0 + CL_; ++t) {
    size_t rowN = (size_t)(b * L_ + t);
    float dt = (float)dt16[rowN * DI_ + d];
    float u  = dt * (float)xc[rowN * DI_ + d];
    const float* BC = dbl + rowN * 96;
    float y = 0.f;
#pragma unroll
    for (int s = 0; s < DS_; ++s) {
      h[s] = h[s] * exp2f(dt * A2[s]) + u * BC[DR_ + s];
      y += h[s] * BC[DR_ + DS_ + s];
    }
    ys[rowN * DI_ + d] = y;
  }
}

// yg = (ys + xc*D) * silu(z)   -> f16
__global__ void k_gate(const float* __restrict__ ys, const _Float16* __restrict__ xc,
                       const _Float16* __restrict__ xz, const float* __restrict__ Dp,
                       _Float16* __restrict__ yg) {
  int i = blockIdx.x * 256 + threadIdx.x;   // < NTOK*DI
  int row = i >> 11, d = i & (DI_ - 1);
  float y = ys[i] + (float)xc[i] * Dp[d];
  float z = (float)xz[(size_t)row * (2 * DI_) + DI_ + d];
  yg[i] = (_Float16)(y * siluf(z));
}

__global__ void k_mul16(const _Float16* __restrict__ a, const _Float16* __restrict__ b,
                        _Float16* __restrict__ o, int n) {
  int i = blockIdx.x * 256 + threadIdx.x;
  if (i < n) o[i] = (_Float16)((float)a[i] * (float)b[i]);
}

// ---- host orchestration ----
extern "C" void kernel_launch(void* const* d_in, const int* in_sizes, int n_in,
                              void* d_out, int out_size, void* d_ws, size_t ws_size,
                              hipStream_t stream) {
  const float* x        = (const float*)d_in[0];
  const float* c        = (const float*)d_in[1];
  const float* adaln_w  = (const float*)d_in[2];
  const float* adaln_b  = (const float*)d_in[3];
  const float* hgd_w1   = (const float*)d_in[4];
  const float* hgd_b1   = (const float*)d_in[5];
  const float* hgd_w2   = (const float*)d_in[6];
  const float* hgd_b2   = (const float*)d_in[7];
  const float* hgf_wm   = (const float*)d_in[8];
  const float* hgf_bm   = (const float*)d_in[9];
  const float* hgf_wr   = (const float*)d_in[10];
  const float* hgf_br   = (const float*)d_in[11];
  const float* hgf_wf   = (const float*)d_in[12];
  const float* hgf_bf   = (const float*)d_in[13];
  const float* in_w     = (const float*)d_in[14];
  const float* conv_w   = (const float*)d_in[15];
  const float* conv_b   = (const float*)d_in[16];
  const float* xproj_w  = (const float*)d_in[17];
  const float* dtproj_w = (const float*)d_in[18];
  const float* dt_bias  = (const float*)d_in[19];
  const float* A_log    = (const float*)d_in[20];
  const float* Dp       = (const float*)d_in[21];
  const float* out_w    = (const float*)d_in[22];
  float* outp = (float*)d_out;

  char* wsb = (char*)d_ws;
  size_t off = 0;
  auto alloc = [&](size_t bytes) -> void* {
    void* p = wsb + off;
    off += (bytes + 255) & ~(size_t)255;
    return p;
  };

  // f16 weights
  _Float16* w16_hgd1   = (_Float16*)alloc((size_t)HR_ * H_ * 2);
  _Float16* w16_hgd2   = (_Float16*)alloc((size_t)H_ * HR_ * 2);
  _Float16* w16_in     = (_Float16*)alloc((size_t)2 * DI_ * H_ * 2);
  _Float16* w16_xproj  = (_Float16*)alloc((size_t)96 * DI_ * 2);
  _Float16* w16_dtproj = (_Float16*)alloc((size_t)DI_ * DR_ * 2);
  _Float16* w16_out    = (_Float16*)alloc((size_t)H_ * DI_ * 2);
  _Float16* w16_wm     = (_Float16*)alloc((size_t)HR_ * H_ * 2);
  _Float16* w16_wr     = (_Float16*)alloc((size_t)HR_ * H_ * 2);
  _Float16* w16_wf     = (_Float16*)alloc((size_t)H_ * HR_ * 2);
  float*    Aneg       = (float*)   alloc((size_t)DI_ * DS_ * 4);
  float*    mod        = (float*)   alloc((size_t)B_ * 3 * H_ * 4);
  // activations
  _Float16* x1h   = (_Float16*)alloc((size_t)NTOK * H_ * 2);
  _Float16* hd1h  = (_Float16*)alloc((size_t)NTOK * HR_ * 2);
  _Float16* hdh   = (_Float16*)alloc((size_t)NTOK * H_ * 2);
  _Float16* xzh   = (_Float16*)alloc((size_t)NTOK * 2 * DI_ * 2);
  _Float16* xch   = (_Float16*)alloc((size_t)NTOK * DI_ * 2);
  float*    dbl   = (float*)   alloc((size_t)NTOK * 96 * 4);
  _Float16* dth   = (_Float16*)alloc((size_t)NTOK * DR_ * 2);
  _Float16* dtf16 = (_Float16*)alloc((size_t)NTOK * DI_ * 2);
  float*    hfin  = (float*)   alloc((size_t)B_ * NC_ * DI_ * DS_ * 4);
  float*    SdtB  = (float*)   alloc((size_t)B_ * NC_ * DI_ * 4);
  float*    h0buf = (float*)   alloc((size_t)B_ * NC_ * DI_ * DS_ * 4);
  float*    ys    = (float*)   alloc((size_t)NTOK * DI_ * 4);
  _Float16* ygh   = (_Float16*)alloc((size_t)NTOK * DI_ * 2);
  _Float16* x12h  = (_Float16*)alloc((size_t)NTOK * H_ * 2);
  _Float16* mh    = (_Float16*)alloc((size_t)NTOK * HR_ * 2);
  _Float16* rh    = (_Float16*)alloc((size_t)NTOK * HR_ * 2);
  _Float16* ph    = (_Float16*)alloc((size_t)NTOK * HR_ * 2);

  auto cvt = [&](const float* src, _Float16* dst, int n) {
    k_cvt16<<<(n + 255) / 256, 256, 0, stream>>>(src, dst, n);
  };
  cvt(hgd_w1,   w16_hgd1,   HR_ * H_);
  cvt(hgd_w2,   w16_hgd2,   H_ * HR_);
  cvt(in_w,     w16_in,     2 * DI_ * H_);
  cvt(xproj_w,  w16_xproj,  96 * DI_);
  cvt(dtproj_w, w16_dtproj, DI_ * DR_);
  cvt(out_w,    w16_out,    H_ * DI_);
  cvt(hgf_wm,   w16_wm,     HR_ * H_);
  cvt(hgf_wr,   w16_wr,     HR_ * H_);
  cvt(hgf_wf,   w16_wf,     H_ * HR_);
  k_negexp<<<(DI_ * DS_ + 255) / 256, 256, 0, stream>>>(A_log, Aneg, DI_ * DS_);

  // adaLN mod + LN/modulate
  k_mod<<<dim3(3 * H_ / 256, B_), 256, 0, stream>>>(c, adaln_w, adaln_b, mod);
  k_ln_mod<<<NTOK, 256, 0, stream>>>(x, mod, x1h);

  const int GM = NTOK / BM;  // 256
  // hourglass dense: silu(x1@w1+b1) -> @w2+b2
  gemm_f16<EPI_BIAS_SILU_F16, false><<<dim3(GM, HR_ / BN), 256, 0, stream>>>(
      x1h, H_, w16_hgd1, H_, HR_, H_, HR_, nullptr, hd1h, hgd_b1, nullptr, nullptr);
  gemm_f16<EPI_BIAS_F16, false><<<dim3(GM, H_ / BN), 256, 0, stream>>>(
      hd1h, HR_, w16_hgd2, HR_, H_, HR_, H_, nullptr, hdh, hgd_b2, nullptr, nullptr);
  // xz = hd @ in_w^T  (the 275-GFLOP GEMM)
  gemm_f16<EPI_F16, false><<<dim3(GM, (2 * DI_) / BN), 256, 0, stream>>>(
      hdh, H_, w16_in, H_, 2 * DI_, H_, 2 * DI_, nullptr, xzh, nullptr, nullptr, nullptr);
  // causal depthwise conv + silu
  k_conv<<<dim3(DI_ / 256, L_, B_), 256, 0, stream>>>(xzh, conv_w, conv_b, xch);
  // dbl = xc @ xproj_w^T   [N,96] f32  (only GEMM needing the N-edge guard)
  gemm_f16<EPI_F32, true><<<dim3(GM, 1), 256, 0, stream>>>(
      xch, DI_, w16_xproj, DI_, 96, DI_, 96, dbl, nullptr, nullptr, nullptr, nullptr);
  k_dth<<<(NTOK * DR_) / 256, 256, 0, stream>>>(dbl, dth, NTOK * DR_);
  // dt = softplus(dt64 @ dtproj^T + dt_bias)  -> f16 [N,DI]
  gemm_f16<EPI_BIAS_SP_F16, false><<<dim3(GM, DI_ / BN), 256, 0, stream>>>(
      dth, DR_, w16_dtproj, DR_, DI_, DR_, DI_, nullptr, dtf16, dt_bias, nullptr, nullptr);
  // chunked selective scan
  k_scan_partial<<<dim3(DI_ / 256, NC_, B_), 256, 0, stream>>>(dtf16, xch, dbl, Aneg, hfin, SdtB);
  k_scan_combine<<<dim3(DI_ / 256, B_), 256, 0, stream>>>(hfin, SdtB, Aneg, h0buf);
  k_scan_apply<<<dim3(DI_ / 256, NC_, B_), 256, 0, stream>>>(dtf16, xch, dbl, Aneg, h0buf, ys);
  // gate: (ys + xc*D) * silu(z)
  k_gate<<<(NTOK * DI_) / 256, 256, 0, stream>>>(ys, xch, xzh, Dp, ygh);
  // x1_2 = yg @ out_w^T
  gemm_f16<EPI_F16, false><<<dim3(GM, H_ / BN), 256, 0, stream>>>(
      ygh, DI_, w16_out, DI_, H_, DI_, H_, nullptr, x12h, nullptr, nullptr, nullptr);
  // hourglass fusion
  gemm_f16<EPI_BIAS_SILU_F16, false><<<dim3(GM, HR_ / BN), 256, 0, stream>>>(
      x12h, H_, w16_wm, H_, HR_, H_, HR_, nullptr, mh, hgf_bm, nullptr, nullptr);
  gemm_f16<EPI_BIAS_SILU_F16, false><<<dim3(GM, HR_ / BN), 256, 0, stream>>>(
      x1h, H_, w16_wr, H_, HR_, H_, HR_, nullptr, rh, hgf_br, nullptr, nullptr);
  k_mul16<<<(NTOK * HR_) / 256, 256, 0, stream>>>(mh, rh, ph, NTOK * HR_);
  // out = x + alpha * (p @ wf^T + bf)
  gemm_f16<EPI_FINAL, false><<<dim3(GM, H_ / BN), 256, 0, stream>>>(
      ph, HR_, w16_wf, HR_, H_, HR_, H_, outp, nullptr, hgf_bf, x, mod);

  (void)in_sizes; (void)n_in; (void)out_size; (void)ws_size;
}
